// MultiHeadSelfAttention_26946624815518
// MI455X (gfx1250) — matrix operand order repaired and checked by reading, not compiled, not run on hardware
//
#include <hip/hip_runtime.h>

// ---------------------------------------------------------------------------
// MI455X (gfx1250, wave32) attention:
//   q=x@Wq, k=x@Wk, v=x@Wv ; E = q k^T * sqrt(64) ; P = softmax(E) ; o = P v ; out = o@Wo
// All contractions via v_wmma_f32_16x16x32_bf16 (bf16 operands, f32 accum).
// Q staging in the attention kernel uses GLOBAL_LOAD_ASYNC_TO_LDS_B128.
// ---------------------------------------------------------------------------

typedef __attribute__((ext_vector_type(16))) __bf16 v16bf;
typedef __attribute__((ext_vector_type(8)))  __bf16 v8bf;
typedef __attribute__((ext_vector_type(4)))  __bf16 v4bf;
typedef __attribute__((ext_vector_type(8)))  float  v8f;

union Frag { v16bf v; v8bf h[2]; };

__device__ __forceinline__ v8f wmma_bf16(const Frag& a, const Frag& b, v8f c) {
  // D = A(16x32) * B(32x16) + C, f32 accum
  return __builtin_amdgcn_wmma_f32_16x16x32_bf16(false, a.v, false, b.v,
                                                 (short)0, c, false, false);
}

// Async global->LDS 16-byte copy (ASYNCcnt-tracked). lds_off = LDS byte offset,
// gaddr = 64-bit global address.  (CDNA5 ISA)
__device__ __forceinline__ void async_g2l_b128(unsigned lds_off, const void* gaddr) {
  asm volatile("global_load_async_to_lds_b128 %0, %1, off"
               :: "v"(lds_off), "v"((unsigned long long)(uintptr_t)gaddr)
               : "memory");
}
__device__ __forceinline__ void wait_asynccnt0() {
  asm volatile("s_wait_asynccnt 0x0" ::: "memory");
}

// ======================= tiled bf16 WMMA GEMM ==============================
// C[M,N] = A[M,K] @ B[K,N].  Block tile 128x128, 8 waves of 32x64, K-step 32.
// Register-prefetch double buffering: tile k+1 is loaded to VGPRs while the
// WMMAs for tile k run from LDS.
// OUT_MODE: 0 = bf16 row-major, 1 = bf16 transposed-per-batch (Vt[b][n][m&4095]),
//           2 = f32 row-major.  ABF16: A operand is bf16 (else f32, converted).
constexpr int TM = 128, TN = 128, TK = 32, LSTR = 40; // LDS row stride (pad 8)

template<int OUT_MODE, bool ABF16>
__global__ __launch_bounds__(256)
void gemm_bf16_wmma(const void* __restrict__ Av, const float* __restrict__ B,
                    void* __restrict__ Cv, int M, int N, int K)
{
  __shared__ __bf16 As[TM * LSTR];
  __shared__ __bf16 Bs[TN * LSTR];   // stored transposed: Bs[n][k]

  const int tid  = threadIdx.x;
  const int lane = tid & 31, wv = tid >> 5;
  const int kh   = lane >> 4, ln = lane & 15;
  const int wm0  = (wv >> 1) * 32, wn0 = (wv & 1) * 64;
  const int m0   = blockIdx.y * TM, n0 = blockIdx.x * TN;

  // B staging map: thread owns column n, a 16-wide k chunk -> packed LDS stores
  const int bn = tid & 127, bk = (tid >> 7) * 16;

  v8f acc[2][4];
  #pragma unroll
  for (int i = 0; i < 2; i++)
    #pragma unroll
    for (int j = 0; j < 4; j++) acc[i][j] = (v8f)(0.0f);

  float4 aF[4];          // f32-A staging regs
  v8bf   aH[2];          // bf16-A staging regs
  float  bR[16];         // B staging regs

  auto load_stage = [&](int k0) {
    if (ABF16) {
      const __bf16* A = (const __bf16*)Av;
      #pragma unroll
      for (int i = 0; i < 2; i++) {
        int e = (tid + i * 256) * 8;
        int row = e >> 5, col = e & 31;
        aH[i] = *(const v8bf*)(A + (size_t)(m0 + row) * K + k0 + col);
      }
    } else {
      const float* A = (const float*)Av;
      #pragma unroll
      for (int i = 0; i < 4; i++) {
        int e = (tid + i * 256) * 4;
        int row = e >> 5, col = e & 31;
        aF[i] = *(const float4*)(A + (size_t)(m0 + row) * K + k0 + col);
      }
    }
    #pragma unroll
    for (int i = 0; i < 16; i++)
      bR[i] = B[(size_t)(k0 + bk + i) * N + n0 + bn];
  };

  auto store_stage = [&]() {
    if (ABF16) {
      #pragma unroll
      for (int i = 0; i < 2; i++) {
        int e = (tid + i * 256) * 8;
        int row = e >> 5, col = e & 31;
        *(v8bf*)&As[row * LSTR + col] = aH[i];
      }
    } else {
      #pragma unroll
      for (int i = 0; i < 4; i++) {
        int e = (tid + i * 256) * 4;
        int row = e >> 5, col = e & 31;
        v4bf p;
        p[0] = (__bf16)aF[i].x; p[1] = (__bf16)aF[i].y;
        p[2] = (__bf16)aF[i].z; p[3] = (__bf16)aF[i].w;
        *(v4bf*)&As[row * LSTR + col] = p;
      }
    }
    v8bf p0, p1;
    #pragma unroll
    for (int i = 0; i < 8; i++) { p0[i] = (__bf16)bR[i]; p1[i] = (__bf16)bR[8 + i]; }
    *(v8bf*)&Bs[bn * LSTR + bk]     = p0;
    *(v8bf*)&Bs[bn * LSTR + bk + 8] = p1;
  };

  load_stage(0);
  for (int k0 = 0; k0 < K; k0 += TK) {
    store_stage();
    __syncthreads();
    if (k0 + TK < K) load_stage(k0 + TK);   // overlap with WMMAs below

    Frag af[2], bfr[4];
    #pragma unroll
    for (int fm = 0; fm < 2; fm++) {
      const __bf16* p = &As[(wm0 + fm * 16 + ln) * LSTR];
      af[fm].h[0] = *(const v8bf*)(p + kh * 16);
      af[fm].h[1] = *(const v8bf*)(p + kh * 16 + 8);
    }
    #pragma unroll
    for (int fn = 0; fn < 4; fn++) {
      const __bf16* p = &Bs[(wn0 + fn * 16 + ln) * LSTR];
      bfr[fn].h[0] = *(const v8bf*)(p + kh * 16);
      bfr[fn].h[1] = *(const v8bf*)(p + kh * 16 + 8);
    }
    #pragma unroll
    for (int fm = 0; fm < 2; fm++)
      #pragma unroll
      for (int fn = 0; fn < 4; fn++)
        acc[fm][fn] = wmma_bf16(af[fm], bfr[fn], acc[fm][fn]);
    __syncthreads();
  }

  // ---- epilogue ----
  #pragma unroll
  for (int fm = 0; fm < 2; fm++)
    #pragma unroll
    for (int fn = 0; fn < 4; fn++) {
      int mbase = m0 + wm0 + fm * 16 + 8 * kh;  // rows mbase..mbase+7
      int n = n0 + wn0 + fn * 16 + ln;
      if (OUT_MODE == 0) {
        __bf16* C = (__bf16*)Cv;
        #pragma unroll
        for (int r = 0; r < 8; r++)
          C[(size_t)(mbase + r) * N + n] = (__bf16)acc[fm][fn][r];
      } else if (OUT_MODE == 1) {
        // Vt[b][n][m&4095]: 8 consecutive m -> one packed b128 store
        __bf16* C = (__bf16*)Cv;
        int b = mbase >> 12, mk = mbase & 4095;
        v8bf pk;
        #pragma unroll
        for (int r = 0; r < 8; r++) pk[r] = (__bf16)acc[fm][fn][r];
        *(v8bf*)(C + ((size_t)b * N + n) * 4096 + mk) = pk;
      } else {
        float* C = (float*)Cv;
        #pragma unroll
        for (int r = 0; r < 8; r++)
          C[(size_t)(mbase + r) * N + n] = acc[fm][fn][r];
      }
    }
}

// =========================== attention kernel ==============================
// One workgroup = 16 query rows. Full score row block E[16][4096] lives in LDS
// (f32, padded stride) -> exact softmax, no online rescaling.
constexpr int   ESTR  = 4100;   // f32 elems per E row (pad 4: conflict-free b128)
constexpr int   QSTR  = 1032;   // bf16 elems per Q row (pad 8)
constexpr float SCALE = 8.0f;   // sqrt(head_dim=64), multiplies energy
constexpr size_t ATTN_SMEM = (size_t)16 * ESTR * 4 + (size_t)16 * QSTR * 2 + 64;

__global__ __launch_bounds__(256)
void attn_kernel(const __bf16* __restrict__ Qb, const __bf16* __restrict__ Kb,
                 const __bf16* __restrict__ Vt, __bf16* __restrict__ Ob)
{
  extern __shared__ char smem[];
  float*  E    = (float*)smem;                                   // [16][ESTR]
  __bf16* Qs   = (__bf16*)(smem + (size_t)16 * ESTR * 4);        // [16][QSTR]
  float*  linv = (float*)(smem + (size_t)16 * ESTR * 4 + (size_t)16 * QSTR * 2);

  const int tid  = threadIdx.x;
  const int lane = tid & 31, wv = tid >> 5;
  const int kh   = lane >> 4, ln = lane & 15;
  const int q0   = blockIdx.x * 16;          // global query row base
  const int b    = q0 >> 12;                 // batch
  const __bf16* Kbb = Kb + (size_t)b * 4096 * 1024;   // K rows of this batch
  const __bf16* Vtb = Vt + (size_t)b * 1024 * 4096;   // Vt[dim][key]

  // ---- phase 0: stage Q tile via async global->LDS (ASYNCcnt) ----
  #pragma unroll
  for (int i = 0; i < 8; i++) {
    int e = (tid + i * 256) * 8;
    int row = e >> 10, col = e & 1023;
    unsigned lds_off = (unsigned)(uintptr_t)&Qs[row * QSTR + col];
    async_g2l_b128(lds_off, Qb + (size_t)(q0 + row) * 1024 + col);
  }
  wait_asynccnt0();
  __syncthreads();

  // ---- phase 1: E[16][4096] = Q @ K^T ----
  // Each wave: 512 keys; two key tiles per A-fragment (2 indep WMMA chains).
  for (int nb = 0; nb < 16; nb++) {
    int n0 = wv * 512 + nb * 32;
    const __bf16* krow0 = Kbb + (size_t)(n0 + ln) * 1024;  // B-frag: K row-major
    const __bf16* krow1 = krow0 + (size_t)16 * 1024;
    v8f c0 = (v8f)(0.0f), c1 = (v8f)(0.0f);
    for (int kb = 0; kb < 32; kb++) {
      int k0 = kb * 32;
      Frag a, b0, b1;
      const __bf16* qp = &Qs[ln * QSTR + k0];
      a.h[0] = *(const v8bf*)(qp + kh * 16);
      a.h[1] = *(const v8bf*)(qp + kh * 16 + 8);
      const __bf16* kp0 = krow0 + k0 + kh * 16;
      b0.h[0] = *(const v8bf*)kp0;
      b0.h[1] = *(const v8bf*)(kp0 + 8);
      const __bf16* kp1 = krow1 + k0 + kh * 16;
      b1.h[0] = *(const v8bf*)kp1;
      b1.h[1] = *(const v8bf*)(kp1 + 8);
      c0 = wmma_bf16(a, b0, c0);
      c1 = wmma_bf16(a, b1, c1);
    }
    #pragma unroll
    for (int r = 0; r < 8; r++) {
      E[(r + 8 * kh) * ESTR + n0 + ln]      = c0[r];
      E[(r + 8 * kh) * ESTR + n0 + 16 + ln] = c1[r];
    }
  }
  __syncthreads();

  // ---- phase 2: softmax over each row (16 lanes per row) ----
  {
    int r = tid >> 4, cc = tid & 15;
    float* Er = E + r * ESTR;
    float mx = -3.4e38f;
    for (int j = cc; j < 4096; j += 16) mx = fmaxf(mx, Er[j]);
    #pragma unroll
    for (int off = 1; off < 16; off <<= 1) mx = fmaxf(mx, __shfl_xor(mx, off, 32));
    mx *= SCALE;
    float sum = 0.0f;
    for (int j = cc; j < 4096; j += 16) {
      float p = __expf(Er[j] * SCALE - mx);
      Er[j] = p;                      // unnormalized prob, f32 in LDS
      sum += p;
    }
    #pragma unroll
    for (int off = 1; off < 16; off <<= 1) sum += __shfl_xor(sum, off, 32);
    if (cc == 0) linv[r] = 1.0f / sum;
  }
  __syncthreads();

  // ---- phase 3: O[16][dim slice 128 per wave] = P @ V ----
  v8f acc[8];
  #pragma unroll
  for (int i = 0; i < 8; i++) acc[i] = (v8f)(0.0f);
  for (int kb = 0; kb < 128; kb++) {
    int k0 = kb * 32;
    const float* ep = &E[ln * ESTR + k0 + kh * 16];
    float4 x0 = *(const float4*)(ep);
    float4 x1 = *(const float4*)(ep + 4);
    float4 x2 = *(const float4*)(ep + 8);
    float4 x3 = *(const float4*)(ep + 12);
    Frag a;   // f32 -> bf16 on the fly
    a.v[0]=(__bf16)x0.x; a.v[1]=(__bf16)x0.y; a.v[2]=(__bf16)x0.z; a.v[3]=(__bf16)x0.w;
    a.v[4]=(__bf16)x1.x; a.v[5]=(__bf16)x1.y; a.v[6]=(__bf16)x1.z; a.v[7]=(__bf16)x1.w;
    a.v[8]=(__bf16)x2.x; a.v[9]=(__bf16)x2.y; a.v[10]=(__bf16)x2.z; a.v[11]=(__bf16)x2.w;
    a.v[12]=(__bf16)x3.x; a.v[13]=(__bf16)x3.y; a.v[14]=(__bf16)x3.z; a.v[15]=(__bf16)x3.w;
    #pragma unroll
    for (int nb = 0; nb < 8; nb++) {
      const __bf16* vp = Vtb + (size_t)(wv * 128 + nb * 16 + ln) * 4096 + k0 + kh * 16;
      Frag bf;
      bf.h[0] = *(const v8bf*)vp;
      bf.h[1] = *(const v8bf*)(vp + 8);
      acc[nb] = wmma_bf16(a, bf, acc[nb]);
    }
  }
  // store O (bf16), folding the 1/rowsum
  #pragma unroll
  for (int nb = 0; nb < 8; nb++)
    #pragma unroll
    for (int r = 0; r < 8; r++) {
      int m = r + 8 * kh;
      float val = acc[nb][r] * linv[m];
      Ob[(size_t)(q0 + m) * 1024 + wv * 128 + nb * 16 + ln] = (__bf16)val;
    }
}

// ============================== launcher ===================================
extern "C" void kernel_launch(void* const* d_in, const int* in_sizes, int n_in,
                              void* d_out, int out_size, void* d_ws, size_t ws_size,
                              hipStream_t stream)
{
  const float* x  = (const float*)d_in[0];
  const float* Wq = (const float*)d_in[1];
  const float* Wk = (const float*)d_in[2];
  const float* Wv = (const float*)d_in[3];
  const float* Wo = (const float*)d_in[4];

  char* ws = (char*)d_ws;
  const size_t SZ = (size_t)8192 * 1024 * 2;       // 16 MB per bf16 buffer
  __bf16* Qb = (__bf16*)(ws + 0 * SZ);             // [8192][1024]
  __bf16* Kb = (__bf16*)(ws + 1 * SZ);             // [8192][1024]
  __bf16* Vt = (__bf16*)(ws + 2 * SZ);             // [2][1024][4096] (dim-major)
  __bf16* Ob = (__bf16*)(ws + 3 * SZ);             // [8192][1024]

  dim3 blk(256);
  dim3 gg(1024 / TN, 8192 / TM);   // (8, 64)

  hipLaunchKernelGGL((gemm_bf16_wmma<0, false>), gg, blk, 0, stream,
                     (const void*)x, Wq, (void*)Qb, 8192, 1024, 1024);
  hipLaunchKernelGGL((gemm_bf16_wmma<0, false>), gg, blk, 0, stream,
                     (const void*)x, Wk, (void*)Kb, 8192, 1024, 1024);
  hipLaunchKernelGGL((gemm_bf16_wmma<1, false>), gg, blk, 0, stream,
                     (const void*)x, Wv, (void*)Vt, 8192, 1024, 1024);

  (void)hipFuncSetAttribute((const void*)attn_kernel,
                            hipFuncAttributeMaxDynamicSharedMemorySize,
                            (int)ATTN_SMEM);
  hipLaunchKernelGGL(attn_kernel, dim3(8192 / 16), blk, ATTN_SMEM, stream,
                     Qb, Kb, Vt, Ob);

  hipLaunchKernelGGL((gemm_bf16_wmma<2, true>), gg, blk, 0, stream,
                     (const void*)Ob, Wo, d_out, 8192, 1024, 1024);
}
